// FAENet_7653631722033
// MI455X (gfx1250) — matrix operand order repaired and checked by reading, not compiled, not run on hardware
//
#include <hip/hip_runtime.h>
#include <hip/hip_bf16.h>

// Problem constants (from reference)
#define NN      50000
#define EE      800000
#define HH      128
#define FF      128
#define GG      50
#define CC      92
#define NGRAPH  64
#define NLAYER  4
#define EPSGN   1e-5f

typedef _Float16 h16;
typedef __attribute__((ext_vector_type(16))) _Float16 v16h;
typedef __attribute__((ext_vector_type(8)))  _Float16 v8h;
typedef __attribute__((ext_vector_type(4)))  _Float16 v4h;
typedef __attribute__((ext_vector_type(8)))  float    v8f;
typedef __attribute__((ext_vector_type(4)))  float    v4f;
typedef __attribute__((ext_vector_type(4)))  int      i32x4;

// Async global->LDS (CDNA5 GLOBAL_LOAD_ASYNC_TO_LDS_B128, ASYNCcnt-tracked),
// guarded so toolchains without the builtins fall back to the sync path.
#if defined(__has_builtin)
#if __has_builtin(__builtin_amdgcn_global_load_async_to_lds_b128) && \
    __has_builtin(__builtin_amdgcn_s_wait_asynccnt)
#define HAVE_ASYNC_LDS 1
#endif
#endif

// Builtin expects int4* operands (global AS1 src, LDS AS3 dst). Go through
// integers so no illegal generic->AS casts are needed; low 32 bits of a
// generic LDS address are the LDS offset (ISA aperture rules).
#define AS1_V4I(p) ((__attribute__((address_space(1))) i32x4*)(unsigned long long)(p))
#define AS3_V4I(p) ((__attribute__((address_space(3))) i32x4*)(unsigned int)(unsigned long long)(p))

__device__ __forceinline__ float swishf(float v) {
    return v / (1.0f + __expf(-v));
}

__device__ __forceinline__ void fatomic_add(float* p, float v) {
    __hip_atomic_fetch_add(p, v, __ATOMIC_RELAXED, __HIP_MEMORY_SCOPE_AGENT);
}

// ---- WMMA fragment loaders (CDNA5 ISA 7.12.2 layouts, wave32) ----
// A: 16x32 f16 tile, row-major source [16 rows][lda], chunks 16B aligned.
// lane<16: M=lane, K=0..7 & 16..23; lane>=16: M=lane-16, K=8..15 & 24..31.
// Two contiguous 16B chunks -> 2x ds_load_b128.
__device__ __forceinline__ v16h ldsA_frag(const h16* base, int lda, int lane) {
    const h16* p = base + (lane & 15) * lda + ((lane >> 4) << 3);
    v8h lo = *(const v8h*)p;
    v8h hi = *(const v8h*)(p + 16);
    return __builtin_shufflevector(lo, hi, 0, 1, 2, 3, 4, 5, 6, 7,
                                           8, 9, 10, 11, 12, 13, 14, 15);
}

// B fragment from a pre-swizzled LDS weight image: per (frag, lane) the 16 halves
// are contiguous (32B) -> 2x ds_load_b128.
__device__ __forceinline__ v16h ldsB_swz(const h16* sWf, int frag, int lane) {
    const h16* p = sWf + (((frag << 5) + lane) << 4);
    v8h lo = *(const v8h*)p;
    v8h hi = *(const v8h*)(p + 8);
    return __builtin_shufflevector(lo, hi, 0, 1, 2, 3, 4, 5, 6, 7,
                                           8, 9, 10, 11, 12, 13, 14, 15);
}

// Stage f32 weight [K=128 x cols=NT*16] into LDS pre-swizzled to B-fragment layout.
// dst[((ks*NT+t)*32+lane)*16+i] = W[(ks*32 + (i>=8?16:0) + ((lane>=16)?8:0) + (i&7)) * cols + t*16 + (lane&15)]
// Rows >= `rows` are zero (K padding, e.g. 92 -> 128).
template <int NT>
__device__ __forceinline__ void stage_weight_swz(h16* dst, const float* W, int rows,
                                                 int tid, int nth) {
    const int cols = NT * 16;
    const int total = 2048 * NT;     // 4 ksteps * NT * 32 lanes * 16
    for (int j = tid; j < total; j += nth) {
        int i    = j & 15;
        int lane = (j >> 4) & 31;
        int frag = j >> 9;
        int t    = frag % NT;
        int ks   = frag / NT;
        int k    = ks * 32 + ((i >> 3) << 4) + ((lane >> 4) << 3) + (i & 7);
        int c    = t * 16 + (lane & 15);
        dst[j] = (k < rows) ? (h16)W[k * cols + c] : (h16)0.f;
    }
}

// Wave-level GEMM: 16-row strip x (NT*16) cols, K=128, f16 in, f32 accumulate.
template <int NT>
__device__ __forceinline__ void wave_gemm_k128(const h16* A, int lda,
                                               const h16* sWf, v8f* acc, int lane) {
#pragma unroll
    for (int ks = 0; ks < 4; ++ks) {
        v16h a = ldsA_frag(A + ks * 32, lda, lane);
#pragma unroll
        for (int t = 0; t < NT; ++t) {
            v16h b = ldsB_swz(sWf, ks * NT + t, lane);
            acc[t] = __builtin_amdgcn_wmma_f32_16x16x32_f16(
                false, a, false, b, (short)0, acc[t], false, false);
        }
    }
}

template <int NT>
__device__ __forceinline__ void init_acc_bias(v8f* acc, const float* bias, int lane) {
#pragma unroll
    for (int t = 0; t < NT; ++t) {
        float bv = bias[t * 16 + (lane & 15)];
#pragma unroll
        for (int r = 0; r < 8; ++r) acc[t][r] = bv;
    }
}

// Write accumulators (optionally swish) back into an LDS f16 strip.
template <int NT>
__device__ __forceinline__ void acc_to_lds(h16* A, int lda, const v8f* acc,
                                           int lane, bool act) {
    const int cb = lane & 15, rb = (lane >> 4) << 3;
#pragma unroll
    for (int t = 0; t < NT; ++t)
#pragma unroll
        for (int r = 0; r < 8; ++r) {
            float v = acc[t][r];
            if (act) v = swishf(v);
            A[(rb + r) * lda + t * 16 + cb] = (h16)v;
        }
}

// Stage f32 rows [srcCols floats each] -> LDS f16 [64][128], zero-padding columns
// >= srcCols and rows >= nMax. float4 loads, v4h (ds_store_b64) stores.
__device__ __forceinline__ void stage_rows_f32(h16* dst, const float* src, int srcCols,
                                               int n0, int nMax, int tid) {
    for (int i = tid; i < 64 * 32; i += 128) {
        int r = i >> 5, c4 = i & 31, node = n0 + r;
        int col = c4 * 4;
        v4h o = {};
        if (node < nMax && col + 3 < srcCols) {
            v4f v = *(const v4f*)(src + (size_t)node * srcCols + col);
            o[0] = (h16)v[0]; o[1] = (h16)v[1]; o[2] = (h16)v[2]; o[3] = (h16)v[3];
        }
        *(v4h*)(dst + r * 128 + col) = o;
    }
}

// -------------------- node embedding: 3 chained GEMMs --------------------
__global__ void __launch_bounds__(128)
node_embed_kernel(const float* __restrict__ x,
                  const float* __restrict__ Wnode, const float* __restrict__ bnode,
                  const float* __restrict__ Wlin,  const float* __restrict__ blin,
                  const float* __restrict__ Wlin2, const float* __restrict__ blin2,
                  float* __restrict__ h) {
    __shared__ alignas(32) h16 sA[64 * 128];
    __shared__ alignas(32) h16 sW[128 * 128];
    const int tid = threadIdx.x, lane = tid & 31, wid = tid >> 5;
    const int n0 = blockIdx.x * 64;

    stage_rows_f32(sA, x, CC, n0, NN, tid);            // x padded K: 92 -> 128
    stage_weight_swz<8>(sW, Wnode, CC, tid, 128);
    __syncthreads();

    v8f acc[8];
    init_acc_bias<8>(acc, bnode, lane);
    wave_gemm_k128<8>(sA + wid * 16 * 128, 128, sW, acc, lane);
    __syncthreads();
    acc_to_lds<8>(sA + wid * 16 * 128, 128, acc, lane, false);   // no swish here
    stage_weight_swz<8>(sW, Wlin, 128, tid, 128);
    __syncthreads();

    init_acc_bias<8>(acc, blin, lane);
    wave_gemm_k128<8>(sA + wid * 16 * 128, 128, sW, acc, lane);
    __syncthreads();
    acc_to_lds<8>(sA + wid * 16 * 128, 128, acc, lane, true);
    stage_weight_swz<8>(sW, Wlin2, 128, tid, 128);
    __syncthreads();

    init_acc_bias<8>(acc, blin2, lane);
    wave_gemm_k128<8>(sA + wid * 16 * 128, 128, sW, acc, lane);
    const int cb = lane & 15, rb = (lane >> 4) << 3;
#pragma unroll
    for (int t = 0; t < 8; ++t)
#pragma unroll
        for (int r = 0; r < 8; ++r) {
            int node = n0 + wid * 16 + rb + r;
            if (node < NN) h[(size_t)node * HH + t * 16 + cb] = swishf(acc[t][r]);
        }
}

// -------------------- edge embedding: geometry + small MLPs + WMMA GEMM --------------------
__global__ void __launch_bounds__(128)
edge_embed_kernel(const float* __restrict__ pos, const int* __restrict__ eidx,
                  const float* __restrict__ We1, const float* __restrict__ be1,
                  const float* __restrict__ We2, const float* __restrict__ be2,
                  const float* __restrict__ We3, const float* __restrict__ be3,
                  h16* __restrict__ eH) {
    __shared__ alignas(32) h16 sE[64 * 128];
    __shared__ alignas(32) h16 sW[128 * 128];
    __shared__ float sG[64 * GG];
    const int tid = threadIdx.x, lane = tid & 31, wid = tid >> 5;
    const int e0 = blockIdx.x * 64;

    stage_weight_swz<8>(sW, We3, 128, tid, 128);

    const int le = tid & 63;
    const int part = tid >> 6;       // 2 threads cooperate on one edge
    const int eg = e0 + le;
    if (eg < EE) {
        int s = eidx[eg], d = eidx[EE + eg];
        float rx = pos[s * 3 + 0] - pos[d * 3 + 0];
        float ry = pos[s * 3 + 1] - pos[d * 3 + 1];
        float rz = pos[s * 3 + 2] - pos[d * 3 + 2];
        if (part == 0) {
            for (int j = 0; j < 64; ++j) {
                float v = rx * We1[j] + ry * We1[64 + j] + rz * We1[128 + j] + be1[j];
                sE[le * 128 + j] = (h16)swishf(v);
            }
        } else {
            float dist = sqrtf(rx * rx + ry * ry + rz * rz + 1e-12f);
            const float delta = 6.0f / 49.0f;
            const float coeff = -0.5f / (delta * delta);
            for (int g = 0; g < GG; ++g) {
                float t = dist - delta * (float)g;
                sG[le * GG + g] = __expf(coeff * t * t);
            }
            for (int j = 0; j < 64; ++j) {
                float v = be2[j];
                for (int g = 0; g < GG; ++g) v += sG[le * GG + g] * We2[g * 64 + j];
                sE[le * 128 + 64 + j] = (h16)swishf(v);
            }
        }
    } else {
        for (int j = 0; j < 64; ++j) sE[le * 128 + part * 64 + j] = (h16)0.f;
    }
    __syncthreads();

    v8f acc[8];
    init_acc_bias<8>(acc, be3, lane);
    wave_gemm_k128<8>(sE + wid * 16 * 128, 128, sW, acc, lane);
    const int cb = lane & 15, rb = (lane >> 4) << 3;
#pragma unroll
    for (int t = 0; t < 8; ++t)
#pragma unroll
        for (int r = 0; r < 8; ++r) {
            int e = e0 + wid * 16 + rb + r;
            if (e < EE) eH[(size_t)e * 128 + t * 16 + cb] = (h16)swishf(acc[t][r]);
        }
}

// -------------------- generic node GEMM -> f16 out (hd = swish(h@Wdown+b)) --------------------
__global__ void __launch_bounds__(128)
node_gemm_f16out_kernel(const float* __restrict__ src, const float* __restrict__ W,
                        const float* __restrict__ b, h16* __restrict__ out) {
    __shared__ alignas(32) h16 sA[64 * 128];
    __shared__ alignas(32) h16 sW[128 * 128];
    const int tid = threadIdx.x, lane = tid & 31, wid = tid >> 5;
    const int n0 = blockIdx.x * 64;
    stage_rows_f32(sA, src, 128, n0, NN, tid);
    stage_weight_swz<8>(sW, W, 128, tid, 128);
    __syncthreads();
    v8f acc[8];
    init_acc_bias<8>(acc, b, lane);
    wave_gemm_k128<8>(sA + wid * 16 * 128, 128, sW, acc, lane);
    const int cb = lane & 15, rb = (lane >> 4) << 3;
#pragma unroll
    for (int t = 0; t < 8; ++t)
#pragma unroll
        for (int r = 0; r < 8; ++r) {
            int node = n0 + wid * 16 + rb + r;
            if (node < NN) out[(size_t)node * 128 + t * 16 + cb] = (h16)swishf(acc[t][r]);
        }
}

// -------------------- fused edge filter GEMM + gather-multiply + scatter-add --------------------
__global__ void __launch_bounds__(128)
edge_msg_kernel(const h16* __restrict__ eH, const float* __restrict__ Wg,
                const float* __restrict__ bg, const h16* __restrict__ hdH,
                const int* __restrict__ eidx, float* __restrict__ agg) {
    __shared__ alignas(32) h16 sE[64 * 128];
    __shared__ alignas(32) h16 sW[128 * 128];
    const int tid = threadIdx.x, lane = tid & 31, wid = tid >> 5;
    const int e0 = blockIdx.x * 64;
    {   // stage the 64x128 f16 e-tile: async DMA straight into LDS when available
        const v8h* s8 = (const v8h*)eH;
        v8h* d8 = (v8h*)sE;
#ifdef HAVE_ASYNC_LDS
        for (int i = tid; i < 64 * 16; i += 128) {
            int le = i >> 4, c8 = i & 15, eg = e0 + le;
            if (eg < EE) {
                __builtin_amdgcn_global_load_async_to_lds_b128(
                    AS1_V4I(s8 + (size_t)eg * 16 + c8), AS3_V4I(d8 + le * 16 + c8),
                    0, 0);
            } else {
                v8h z = {};
                d8[le * 16 + c8] = z;
            }
        }
        __builtin_amdgcn_s_wait_asynccnt(0);   // drain this wave's DMAs pre-barrier
#else
        for (int i = tid; i < 64 * 16; i += 128) {
            int le = i >> 4, c8 = i & 15, eg = e0 + le;
            v8h v = {};
            if (eg < EE) v = s8[(size_t)eg * 16 + c8];
            d8[le * 16 + c8] = v;
        }
#endif
    }
    stage_weight_swz<8>(sW, Wg, 128, tid, 128);
    __syncthreads();

    v8f acc[8];
    init_acc_bias<8>(acc, bg, lane);
    wave_gemm_k128<8>(sE + wid * 16 * 128, 128, sW, acc, lane);

    const int cb = lane & 15, rb = (lane >> 4) << 3;
    int sArr[8], dArr[8];
#pragma unroll
    for (int r = 0; r < 8; ++r) {
        int e = e0 + wid * 16 + rb + r;
        sArr[r] = (e < EE) ? eidx[e] : -1;
        dArr[r] = (e < EE) ? eidx[EE + e] : -1;
    }
#pragma unroll
    for (int t = 0; t < 8; ++t) {
        int col = t * 16 + cb;
#pragma unroll
        for (int r = 0; r < 8; ++r) {
            if (sArr[r] >= 0) {
                float w = swishf(acc[t][r]);                            // W = swish(e@Wgeom+b)
                float m = w * (float)hdH[(size_t)sArr[r] * 128 + col];  // msg = hd[src]*W
                fatomic_add(&agg[(size_t)dArr[r] * 128 + col], m);      // segsum over dst
            }
        }
    }
}

// -------------------- GraphNorm stats (batch is sorted -> run-length reduce) ----------
__global__ void cnt_kernel(const int* __restrict__ batch, float* __restrict__ cnt) {
    int i = blockIdx.x * 256 + threadIdx.x;
    if (i < NN) fatomic_add(&cnt[batch[i]], 1.f);
}

// One thread per column; walks 32 consecutive (sorted) nodes, one atomic per run.
__global__ void __launch_bounds__(128)
gn_mean_kernel(const float* __restrict__ agg, const int* __restrict__ batch,
               float* __restrict__ meanS) {
    const int c = threadIdx.x;
    const int n0 = blockIdx.x * 32;
    float acc = 0.f;
    int curg = -1;
    for (int r = 0; r < 32; ++r) {
        int node = n0 + r;
        if (node >= NN) break;
        int g = batch[node];
        if (g != curg) {
            if (curg >= 0) fatomic_add(&meanS[curg * HH + c], acc);
            curg = g; acc = 0.f;
        }
        acc += agg[(size_t)node * HH + c];
    }
    if (curg >= 0) fatomic_add(&meanS[curg * HH + c], acc);
}

__global__ void __launch_bounds__(128)
gn_var_kernel(const float* __restrict__ agg, const int* __restrict__ batch,
              const float* __restrict__ meanS, const float* __restrict__ cnt,
              const float* __restrict__ gnms_l, float* __restrict__ varS) {
    const int c = threadIdx.x;
    const int n0 = blockIdx.x * 32;
    const float ms = gnms_l[c];
    float acc = 0.f, m = 0.f;
    int curg = -1;
    for (int r = 0; r < 32; ++r) {
        int node = n0 + r;
        if (node >= NN) break;
        int g = batch[node];
        if (g != curg) {
            if (curg >= 0) fatomic_add(&varS[curg * HH + c], acc);
            curg = g; acc = 0.f;
            m = ms * (meanS[g * HH + c] / fmaxf(cnt[g], 1.f));
        }
        float ctr = agg[(size_t)node * HH + c] - m;
        acc += ctr * ctr;
    }
    if (curg >= 0) fatomic_add(&varS[curg * HH + c], acc);
}

__global__ void __launch_bounds__(128)
gn_up_kernel(const float* __restrict__ agg, const int* __restrict__ batch,
             const float* __restrict__ meanS, const float* __restrict__ varS,
             const float* __restrict__ cnt, const float* __restrict__ gnms_l,
             const float* __restrict__ gnw_l, const float* __restrict__ gnb_l,
             const float* __restrict__ Wup_l, const float* __restrict__ bup_l,
             float* __restrict__ h) {
    __shared__ alignas(32) h16 sA[64 * 128];
    __shared__ alignas(32) h16 sW[128 * 128];
    const int tid = threadIdx.x, lane = tid & 31, wid = tid >> 5;
    const int n0 = blockIdx.x * 64;
    for (int i = tid; i < 64 * 128; i += 128) {
        int r = i >> 7, c = i & 127, node = n0 + r;
        float v = 0.f;
        if (node < NN) {
            int g = batch[node];
            float cg = fmaxf(cnt[g], 1.f);
            float m = meanS[g * HH + c] / cg;
            float ctr = agg[(size_t)node * HH + c] - gnms_l[c] * m;
            float var = varS[g * HH + c] / cg;
            float hn = gnw_l[c] * ctr * rsqrtf(var + EPSGN) + gnb_l[c];
            v = swishf(hn);
        }
        sA[i] = (h16)v;
    }
    stage_weight_swz<8>(sW, Wup_l, 128, tid, 128);
    __syncthreads();
    v8f acc[8];
    init_acc_bias<8>(acc, bup_l, lane);
    wave_gemm_k128<8>(sA + wid * 16 * 128, 128, sW, acc, lane);
    const int cb = lane & 15, rb = (lane >> 4) << 3;
#pragma unroll
    for (int t = 0; t < 8; ++t)
#pragma unroll
        for (int r = 0; r < 8; ++r) {
            int node = n0 + wid * 16 + rb + r;
            if (node < NN) {
                size_t o = (size_t)node * HH + t * 16 + cb;
                h[o] = h[o] + swishf(acc[t][r]);    // residual
            }
        }
}

// -------------------- output MLP + per-graph pooling --------------------
__global__ void __launch_bounds__(128)
output_kernel(const float* __restrict__ h, const int* __restrict__ batch,
              const float* __restrict__ Wo1, const float* __restrict__ bo1,
              const float* __restrict__ Wo2, const float* __restrict__ bo2,
              float* __restrict__ out) {
    __shared__ alignas(32) h16   sA[64 * 128];
    __shared__ alignas(32) h16   sW[128 * 64];
    __shared__ float sO[64 * 64];
    const int tid = threadIdx.x, lane = tid & 31, wid = tid >> 5;
    const int n0 = blockIdx.x * 64;
    stage_rows_f32(sA, h, 128, n0, NN, tid);
    stage_weight_swz<4>(sW, Wo1, 128, tid, 128);
    __syncthreads();
    v8f acc[4];
    init_acc_bias<4>(acc, bo1, lane);
    wave_gemm_k128<4>(sA + wid * 16 * 128, 128, sW, acc, lane);
    const int cb = lane & 15, rb = (lane >> 4) << 3;
#pragma unroll
    for (int t = 0; t < 4; ++t)
#pragma unroll
        for (int r = 0; r < 8; ++r)
            sO[(wid * 16 + rb + r) * 64 + t * 16 + cb] = swishf(acc[t][r]);
    __syncthreads();
    if (tid < 64) {
        int node = n0 + tid;
        if (node < NN) {
            float s = bo2[0];
            for (int c = 0; c < 64; ++c) s += sO[tid * 64 + c] * Wo2[c];
            fatomic_add(&out[batch[node]], s);
        }
    }
}

// -------------------- host orchestration --------------------
extern "C" void kernel_launch(void* const* d_in, const int* in_sizes, int n_in,
                              void* d_out, int out_size, void* d_ws, size_t ws_size,
                              hipStream_t stream) {
    const float* x     = (const float*)d_in[0];
    const float* pos   = (const float*)d_in[1];
    const int*   eidx  = (const int*)d_in[2];
    const int*   batch = (const int*)d_in[3];
    const float* Wnode = (const float*)d_in[4];  const float* bnode = (const float*)d_in[5];
    const float* Wlin  = (const float*)d_in[6];  const float* blin  = (const float*)d_in[7];
    const float* Wlin2 = (const float*)d_in[8];  const float* blin2 = (const float*)d_in[9];
    const float* We1   = (const float*)d_in[10]; const float* be1   = (const float*)d_in[11];
    const float* We2   = (const float*)d_in[12]; const float* be2   = (const float*)d_in[13];
    const float* We3   = (const float*)d_in[14]; const float* be3   = (const float*)d_in[15];
    const float* Wgeom = (const float*)d_in[16]; const float* bgeom = (const float*)d_in[17];
    const float* Wdown = (const float*)d_in[18]; const float* bdown = (const float*)d_in[19];
    const float* Wup   = (const float*)d_in[20]; const float* bup   = (const float*)d_in[21];
    const float* gnw   = (const float*)d_in[22]; const float* gnb   = (const float*)d_in[23];
    const float* gnms  = (const float*)d_in[24];
    const float* Wo1   = (const float*)d_in[25]; const float* bo1   = (const float*)d_in[26];
    const float* Wo2   = (const float*)d_in[27]; const float* bo2   = (const float*)d_in[28];
    float* out = (float*)d_out;

    char* ws = (char*)d_ws;
    size_t off = 0;
    auto alloc = [&](size_t bytes) -> void* {
        void* p = ws + off;
        off = (off + bytes + 255) & ~(size_t)255;
        return p;
    };
    h16*   eH    = (h16*)  alloc((size_t)EE * FF * sizeof(h16));
    h16*   hdH   = (h16*)  alloc((size_t)NN * FF * sizeof(h16));
    float* h     = (float*)alloc((size_t)NN * HH * sizeof(float));
    float* agg   = (float*)alloc((size_t)NN * HH * sizeof(float));
    float* meanS = (float*)alloc((size_t)NGRAPH * HH * sizeof(float));
    float* varS  = (float*)alloc((size_t)NGRAPH * HH * sizeof(float));
    float* cnt   = (float*)alloc((size_t)NGRAPH * sizeof(float));

    (void)hipMemsetAsync(out, 0, NGRAPH * sizeof(float), stream);
    (void)hipMemsetAsync(cnt, 0, NGRAPH * sizeof(float), stream);
    cnt_kernel<<<(NN + 255) / 256, 256, 0, stream>>>(batch, cnt);

    node_embed_kernel<<<(NN + 63) / 64, 128, 0, stream>>>(
        x, Wnode, bnode, Wlin, blin, Wlin2, blin2, h);
    edge_embed_kernel<<<(EE + 63) / 64, 128, 0, stream>>>(
        pos, eidx, We1, be1, We2, be2, We3, be3, eH);

    for (int l = 0; l < NLAYER; ++l) {
        node_gemm_f16out_kernel<<<(NN + 63) / 64, 128, 0, stream>>>(
            h, Wdown + (size_t)l * HH * FF, bdown + l * FF, hdH);
        (void)hipMemsetAsync(agg, 0, (size_t)NN * HH * sizeof(float), stream);
        (void)hipMemsetAsync(meanS, 0, NGRAPH * HH * sizeof(float), stream);
        (void)hipMemsetAsync(varS, 0, NGRAPH * HH * sizeof(float), stream);
        edge_msg_kernel<<<(EE + 63) / 64, 128, 0, stream>>>(
            eH, Wgeom + (size_t)l * FF * FF, bgeom + l * FF, hdH, eidx, agg);
        gn_mean_kernel<<<(NN + 31) / 32, 128, 0, stream>>>(agg, batch, meanS);
        gn_var_kernel<<<(NN + 31) / 32, 128, 0, stream>>>(
            agg, batch, meanS, cnt, gnms + l * HH, varS);
        gn_up_kernel<<<(NN + 63) / 64, 128, 0, stream>>>(
            agg, batch, meanS, varS, cnt, gnms + l * HH, gnw + l * HH, gnb + l * HH,
            Wup + (size_t)l * FF * HH, bup + l * HH, h);
    }

    output_kernel<<<(NN + 63) / 64, 128, 0, stream>>>(h, batch, Wo1, bo1, Wo2, bo2, out);
}
